// GraphIsomorphismLayer_21887153341120
// MI455X (gfx1250) — matrix operand order repaired and checked by reading, not compiled, not run on hardware
//
#include <hip/hip_runtime.h>
#include <hip/hip_bf16.h>

// Problem constants (match reference)
#define NN     50000
#define EE     800000
#define D_IN   96
#define D_OUT  256
#define BN_EPS 1e-3f

typedef __attribute__((ext_vector_type(16))) __bf16 v16bf;
typedef __attribute__((ext_vector_type(8)))  __bf16 v8bf;
typedef __attribute__((ext_vector_type(8)))  float  v8f;

// ---------- helpers ----------
__device__ __forceinline__ unsigned short bf16_bits(__bf16 b) {
    union { __bf16 b; unsigned short s; } c; c.b = b;
    return c.s;
}
__device__ __forceinline__ unsigned short f32_to_bf16_bits(float f) {
    return bf16_bits(static_cast<__bf16>(f));   // native cvt (RNE) on CDNA5
}

// gfx1250 async global->LDS 16B copy (per active lane), tracked by ASYNCcnt.
__device__ __forceinline__ void async_g2l_b128(unsigned lds_off, unsigned goff,
                                               const void* gbase) {
    asm volatile("global_load_async_to_lds_b128 %0, %1, %2"
                 :: "v"(lds_off), "v"(goff), "s"(gbase) : "memory");
}
__device__ __forceinline__ void wait_async0() {
    asm volatile("s_wait_asynccnt 0" ::: "memory");
}

// ---------- kernel 1: agg = eps * x ----------
__global__ void init_agg(const float* __restrict__ x, const float* __restrict__ eps,
                         float* __restrict__ agg) {
    long i = (long)blockIdx.x * blockDim.x + threadIdx.x;
    float e = eps[0];
    if (i < (long)NN * D_IN) agg[i] = e * x[i];
}

// ---------- kernel 2: edge scatter (one wave32 per edge, 3 feats/lane) ----------
__global__ void scatter_edges(const float* __restrict__ x,
                              const int* __restrict__ src,
                              const int* __restrict__ dst,
                              const float* __restrict__ vals,
                              float* __restrict__ agg) {
    int wave = (blockIdx.x * blockDim.x + threadIdx.x) >> 5;
    int lane = threadIdx.x & 31;
    if (wave >= EE) return;
    int   s = src[wave];
    int   d = dst[wave];
    float v = vals[wave];
    const float* xr = x + (long)s * D_IN;
    float*       ar = agg + (long)d * D_IN;
#pragma unroll
    for (int f = 0; f < 3; ++f) {
        int c = lane + 32 * f;
        __hip_atomic_fetch_add(&ar[c], v * xr[c],
                               __ATOMIC_RELAXED, __HIP_MEMORY_SCOPE_AGENT);
    }
}

// ---------- kernel 3: pack W (f32, KxN row-major) -> bf16 fragment-major ----------
// Wp[((kt*16 + nt)*32 + lane)*16 + j] = bf16(W[(kt*32 + (lane>>4)*16 + j)*256 + nt*16 + (lane&15)])
// B-fragment element j has k = khalf*16 + j, so each lane's fragment is one
// contiguous 32B chunk -> two global_load_b128 per fragment at GEMM time.
__global__ void pack_weights(const float* __restrict__ W, unsigned short* __restrict__ Wp,
                             int total) {
    int i = blockIdx.x * blockDim.x + threadIdx.x;
    if (i >= total) return;
    int j    = i & 15;
    int lane = (i >> 4) & 31;
    int ntkt = i >> 9;
    int nt   = ntkt & 15;
    int kt   = ntkt >> 4;
    int n = nt * 16 + (lane & 15);
    int k = kt * 32 + (lane >> 4) * 16 + j;
    Wp[i] = f32_to_bf16_bits(W[(long)k * D_OUT + n]);
}

// ---------- kernel 4: h = relu(bn(agg @ W0)) -> bf16, row-major ----------
// One block = 16-row M panel staged to LDS (async), 8 waves x 2 column tiles.
__global__ void __launch_bounds__(256)
gemm0_bn_relu(const float* __restrict__ agg,
              const unsigned short* __restrict__ W0p,
              const float* __restrict__ gamma,
              const float* __restrict__ beta,
              const float* __restrict__ mean,
              const float* __restrict__ var,
              unsigned short* __restrict__ hbf) {
    __shared__ alignas(16) float shA[16 * D_IN];            // 6 KB fp32 panel

    int tid   = threadIdx.x;
    int mtile = blockIdx.x;

    // Stage A panel: 16 rows x 96 f32 = 6144 B = 384 x 16B chunks, async to LDS.
    const float* Abase = agg + (long)mtile * 16 * D_IN;
    unsigned lds_base = (unsigned)(unsigned long long)(void*)shA;
    {
        unsigned c0 = tid * 16u;
        async_g2l_b128(lds_base + c0, c0, Abase);           // chunks 0..255
        if (tid < 128) {
            unsigned c1 = (tid + 256) * 16u;
            async_g2l_b128(lds_base + c1, c1, Abase);       // chunks 256..383
        }
    }
    wait_async0();
    __syncthreads();

    int wave = tid >> 5, lane = tid & 31;
    int l16 = lane & 15, khi = lane >> 4;
    int nt0 = wave * 2, nt1 = nt0 + 1;

    v8f c0 = {}, c1 = {};
#pragma unroll
    for (int k0 = 0; k0 < D_IN; k0 += 32) {
        int kt = k0 >> 5;
        // A fragment: one contiguous 16-float run per lane, from LDS (ds_load_b128 x4)
        const float* ar = &shA[l16 * D_IN + k0 + 16 * khi];
        float4 a0 = *(const float4*)(ar);
        float4 a1 = *(const float4*)(ar + 4);
        float4 a2 = *(const float4*)(ar + 8);
        float4 a3 = *(const float4*)(ar + 12);
        v16bf a;
        a[0]=(__bf16)a0.x; a[1]=(__bf16)a0.y; a[2]=(__bf16)a0.z; a[3]=(__bf16)a0.w;
        a[4]=(__bf16)a1.x; a[5]=(__bf16)a1.y; a[6]=(__bf16)a1.z; a[7]=(__bf16)a1.w;
        a[8]=(__bf16)a2.x; a[9]=(__bf16)a2.y; a[10]=(__bf16)a2.z; a[11]=(__bf16)a2.w;
        a[12]=(__bf16)a3.x; a[13]=(__bf16)a3.y; a[14]=(__bf16)a3.z; a[15]=(__bf16)a3.w;

        // B fragments: one contiguous 32B per-lane chunk each (global_load_b128 x2)
        const v8bf* bp0 = (const v8bf*)(W0p + (((long)(kt * 16 + nt0) * 32 + lane) * 16));
        const v8bf* bp1 = (const v8bf*)(W0p + (((long)(kt * 16 + nt1) * 32 + lane) * 16));
        v16bf b0 = __builtin_shufflevector(bp0[0], bp0[1], 0,1,2,3,4,5,6,7,8,9,10,11,12,13,14,15);
        v16bf b1 = __builtin_shufflevector(bp1[0], bp1[1], 0,1,2,3,4,5,6,7,8,9,10,11,12,13,14,15);

        c0 = __builtin_amdgcn_wmma_f32_16x16x32_bf16(false, a, false, b0, (short)0, c0, false, false);
        c1 = __builtin_amdgcn_wmma_f32_16x16x32_bf16(false, a, false, b1, (short)0, c1, false, false);
    }

    // Fused BN + ReLU epilogue; C/D layout: lane -> N col, VGPR r -> M = r + 8*khi
#pragma unroll
    for (int p = 0; p < 2; ++p) {
        int bcol = (p ? nt1 : nt0) * 16 + l16;
        v8f cc = p ? c1 : c0;
        float sc = gamma[bcol] * __frsqrt_rn(var[bcol] + BN_EPS);
        float sh = beta[bcol] - mean[bcol] * sc;
#pragma unroll
        for (int r = 0; r < 8; ++r) {
            int m = mtile * 16 + r + 8 * khi;
            float h = fmaxf(cc[r] * sc + sh, 0.0f);
            hbf[(long)m * D_OUT + bcol] = f32_to_bf16_bits(h);
        }
    }
}

// ---------- kernel 5: out = h @ W1 ----------
__global__ void __launch_bounds__(256)
gemm1(const unsigned short* __restrict__ hbf,
      const unsigned short* __restrict__ W1p,
      float* __restrict__ out) {
    __shared__ alignas(16) unsigned short shH[16 * D_OUT];  // 8 KB bf16 panel

    int tid   = threadIdx.x;
    int mtile = blockIdx.x;

    // Stage A panel: 16 rows x 256 bf16 = 8192 B = 512 x 16B chunks, async to LDS.
    const unsigned short* Hbase = hbf + (long)mtile * 16 * D_OUT;
    unsigned lds_base = (unsigned)(unsigned long long)(void*)shH;
    {
        unsigned c0 = tid * 16u;
        async_g2l_b128(lds_base + c0, c0, Hbase);
        unsigned c1 = (tid + 256) * 16u;
        async_g2l_b128(lds_base + c1, c1, Hbase);
    }
    wait_async0();
    __syncthreads();

    int wave = tid >> 5, lane = tid & 31;
    int l16 = lane & 15, khi = lane >> 4;
    int nt0 = wave * 2, nt1 = nt0 + 1;

    v8f c0 = {}, c1 = {};
#pragma unroll
    for (int k0 = 0; k0 < D_OUT; k0 += 32) {
        int kt = k0 >> 5;
        // A fragment: one contiguous 16-bf16 run per lane from LDS (ds_load_b128 x2)
        const unsigned short* ap = &shH[l16 * D_OUT + k0 + 16 * khi];
        v8bf alo = *(const v8bf*)(ap);
        v8bf ahi = *(const v8bf*)(ap + 8);
        v16bf a = __builtin_shufflevector(alo, ahi, 0,1,2,3,4,5,6,7,8,9,10,11,12,13,14,15);

        const v8bf* bp0 = (const v8bf*)(W1p + (((long)(kt * 16 + nt0) * 32 + lane) * 16));
        const v8bf* bp1 = (const v8bf*)(W1p + (((long)(kt * 16 + nt1) * 32 + lane) * 16));
        v16bf b0 = __builtin_shufflevector(bp0[0], bp0[1], 0,1,2,3,4,5,6,7,8,9,10,11,12,13,14,15);
        v16bf b1 = __builtin_shufflevector(bp1[0], bp1[1], 0,1,2,3,4,5,6,7,8,9,10,11,12,13,14,15);

        c0 = __builtin_amdgcn_wmma_f32_16x16x32_bf16(false, a, false, b0, (short)0, c0, false, false);
        c1 = __builtin_amdgcn_wmma_f32_16x16x32_bf16(false, a, false, b1, (short)0, c1, false, false);
    }

#pragma unroll
    for (int p = 0; p < 2; ++p) {
        int bcol = (p ? nt1 : nt0) * 16 + l16;
        v8f cc = p ? c1 : c0;
#pragma unroll
        for (int r = 0; r < 8; ++r) {
            int m = mtile * 16 + r + 8 * khi;
            out[(long)m * D_OUT + bcol] = cc[r];
        }
    }
}

// ---------- launcher ----------
extern "C" void kernel_launch(void* const* d_in, const int* in_sizes, int n_in,
                              void* d_out, int out_size, void* d_ws, size_t ws_size,
                              hipStream_t stream) {
    (void)in_sizes; (void)n_in; (void)out_size; (void)ws_size;

    const float* x        = (const float*)d_in[0];
    const int*   adj_src  = (const int*)d_in[1];
    const int*   adj_dst  = (const int*)d_in[2];
    const float* adj_vals = (const float*)d_in[3];
    const float* eps      = (const float*)d_in[4];
    const float* W0       = (const float*)d_in[5];
    const float* W1       = (const float*)d_in[6];
    const float* gamma    = (const float*)d_in[7];
    const float* beta     = (const float*)d_in[8];
    const float* bn_mean  = (const float*)d_in[9];
    const float* bn_var   = (const float*)d_in[10];
    float*       out      = (float*)d_out;

    // Workspace: agg f32 [N,96] | h bf16 [N,256] | W0p bf16 | W1p bf16  (~45 MB)
    char* ws = (char*)d_ws;
    float*          agg = (float*)ws;
    size_t off = (size_t)NN * D_IN * sizeof(float);
    unsigned short* hbf = (unsigned short*)(ws + off);
    off += (size_t)NN * D_OUT * sizeof(unsigned short);
    unsigned short* W0p = (unsigned short*)(ws + off);
    off += (size_t)D_IN * D_OUT * sizeof(unsigned short);
    unsigned short* W1p = (unsigned short*)(ws + off);

    {   // 1) agg = eps * x
        long total = (long)NN * D_IN;
        init_agg<<<(int)((total + 255) / 256), 256, 0, stream>>>(x, eps, agg);
    }
    {   // 2) edge scatter, 8 waves (edges) / block
        scatter_edges<<<(EE + 7) / 8, 256, 0, stream>>>(x, adj_src, adj_dst, adj_vals, agg);
    }
    {   // 3) weight packing (fragment-major bf16)
        pack_weights<<<(D_IN * D_OUT + 255) / 256, 256, 0, stream>>>(W0, W0p, D_IN * D_OUT);
        pack_weights<<<(D_OUT * D_OUT + 255) / 256, 256, 0, stream>>>(W1, W1p, D_OUT * D_OUT);
    }
    {   // 4) GEMM0 + BN + ReLU (3125 M-panels)
        gemm0_bn_relu<<<NN / 16, 256, 0, stream>>>(agg, W0p, gamma, beta, bn_mean, bn_var, hbf);
    }
    {   // 5) GEMM1 (3125 M-panels)
        gemm1<<<NN / 16, 256, 0, stream>>>(hbf, W1p, out);
    }
}
